// GATLayer_53214644797468
// MI455X (gfx1250) — matrix operand order repaired and checked by reading, not compiled, not run on hardware
//
#include <hip/hip_runtime.h>
#include <hip/hip_bf16.h>

// GAT layer fused kernels for gfx1250 (MI455X), wave32 + WMMA bf16 + async LDS.

#define N_NODES 8192
#define F_IN    512
#define F_OUT   128
#define ALPHA   0.2f
#define MASK_VAL (-9.0e15f)
#define JT      64                      // j-tile width in fused kernel

typedef __attribute__((ext_vector_type(16))) __bf16 v16bf;
typedef __attribute__((ext_vector_type(8)))  float  v8f;

union AFrag { v16bf v; unsigned int u[8]; };
union CFrag { v8f  v;  float f[8]; };

// fp32 -> bf16 round-to-nearest-even
static __device__ __forceinline__ unsigned short f2bf(float x) {
  unsigned int u = __float_as_uint(x);
  unsigned int r = (u + 0x7fffu + ((u >> 16) & 1u)) >> 16;
  return (unsigned short)r;
}

// LDS byte offset of a generic pointer to __shared__ (low 32 bits per ISA).
static __device__ __forceinline__ unsigned lds_off(const void* p) {
  return (unsigned)(unsigned long long)p;
}

// Async global->LDS 16-byte copy (per-lane), tracked by ASYNCcnt.
static __device__ __forceinline__ void async_copy_b128(unsigned laddr,
                                                       const void* gptr) {
  unsigned long long gaddr = (unsigned long long)gptr;
  asm volatile("global_load_async_to_lds_b128 %0, %1, off"
               :: "v"(laddr), "v"(gaddr) : "memory");
}

// ---------------------------------------------------------------------------
// Kernel 1: Wh = h @ w  (bf16 WMMA, f32 accumulate). Also emits bf16 copy.
// Block: 256 threads = 8 waves. Block owns 16 rows; wave w owns 16-col tile w.
// ---------------------------------------------------------------------------
__global__ void __launch_bounds__(256) wh_gemm_kernel(
    const float* __restrict__ h, const float* __restrict__ w,
    float* __restrict__ Wh, unsigned short* __restrict__ Whb) {
  const int lane = threadIdx.x & 31;
  const int wv   = threadIdx.x >> 5;       // n-tile 0..7
  const int i0   = blockIdx.x * 16;
  const int mL   = lane & 15;              // A row / B col / C col
  const int hi   = lane >> 4;              // lane half

  const int n = wv * 16 + mL;              // output column this lane owns

  CFrag c;
  #pragma unroll
  for (int v = 0; v < 8; ++v) c.f[v] = 0.0f;

  for (int k0 = 0; k0 < F_IN; k0 += 32) {
    AFrag a, b;
    #pragma unroll
    for (int v = 0; v < 8; ++v) {
      // A (16x32 bf16): VGPR v holds K pair; lanes 16-31 shifted by +8.
      int ka = (v < 4) ? (2 * v + 8 * hi) : (16 + 2 * (v - 4) + 8 * hi);
      float2 av = *(const float2*)(h + (size_t)(i0 + mL) * F_IN + k0 + ka);
      a.u[v] = (unsigned int)f2bf(av.x) | ((unsigned int)f2bf(av.y) << 16);
      // B (32x16 bf16): VGPR v holds K pair; lanes 16-31 shifted by +8 (same as A).
      int kb = (v < 4) ? (2 * v + 8 * hi) : (16 + 2 * (v - 4) + 8 * hi);
      float b0 = w[(size_t)(k0 + kb) * F_OUT + n];
      float b1 = w[(size_t)(k0 + kb + 1) * F_OUT + n];
      b.u[v] = (unsigned int)f2bf(b0) | ((unsigned int)f2bf(b1) << 16);
    }
    c.v = __builtin_amdgcn_wmma_f32_16x16x32_bf16(false, a.v, false, b.v,
                                                  (short)0, c.v, false, false);
  }

  // C/D layout: VGPR v -> row v (lanes 0-15) or v+8 (lanes 16-31), col = lane%16
  #pragma unroll
  for (int v = 0; v < 8; ++v) {
    int row = i0 + v + 8 * hi;
    float val = c.f[v];
    Wh[(size_t)row * F_OUT + n]  = val;
    Whb[(size_t)row * F_OUT + n] = f2bf(val);
  }
}

// ---------------------------------------------------------------------------
// Kernel 2: Wh1[i] = Wh[i,:]·a[:128];  Wh2[i] = Wh[i,:]·a[128:]
// ---------------------------------------------------------------------------
__global__ void __launch_bounds__(256) attvec_kernel(
    const float* __restrict__ Wh, const float* __restrict__ a,
    float* __restrict__ Wh1, float* __restrict__ Wh2) {
  int row = blockIdx.x * blockDim.x + threadIdx.x;
  if (row >= N_NODES) return;
  const float* p = Wh + (size_t)row * F_OUT;
  float s1 = 0.0f, s2 = 0.0f;
  #pragma unroll 8
  for (int n = 0; n < F_OUT; ++n) {
    float v = p[n];
    s1 += v * a[n];
    s2 += v * a[F_OUT + n];
  }
  Wh1[row] = s1;
  Wh2[row] = s2;
}

// ---------------------------------------------------------------------------
// Kernel 3: fused masked-softmax attention matmul (flash-style online softmax)
// with async double-buffered LDS staging of adj and Wh(bf16) tiles.
// Block: 256 threads = 8 waves, owns 16 output rows; loops j in 64-col tiles.
// out[i,:] = elu( softmax_row(mask(leakyrelu(Wh1_i + Wh2_j), adj)) @ Wh )
// ---------------------------------------------------------------------------
__global__ void __launch_bounds__(256) gat_fused_kernel(
    const int* __restrict__ adj, const unsigned short* __restrict__ Whb,
    const float* __restrict__ Wh1, const float* __restrict__ Wh2,
    float* __restrict__ out) {
  __shared__ __align__(16) int            adjt[2][16][JT];    // 2 x 4 KB
  __shared__ __align__(16) unsigned short whbt[2][JT][128];   // 2 x 16 KB
  __shared__ __align__(8)  unsigned short ptile[16][JT];      // 2 KB (bf16 p)
  __shared__ float pmax[16][16];                              // 1 KB
  __shared__ float mrow[16], lrow[16], scale_s[16], mnew_s[16], rsum_s[16];

  const int lane = threadIdx.x & 31;
  const int wv   = threadIdx.x >> 5;
  const int i0   = blockIdx.x * 16;
  const int mL   = lane & 15;
  const int hi   = lane >> 4;

  if (threadIdx.x < 16) { mrow[threadIdx.x] = -3.0e38f; lrow[threadIdx.x] = 0.0f; }

  // score-tile mapping: thread handles e[em][ej..ej+3]
  const int em = threadIdx.x >> 4;          // 0..15
  const int ej = (threadIdx.x & 15) * 4;    // 0,4,...,60
  const float wh1_m = Wh1[i0 + em];

  // per-thread staging chunk for the Wh bf16 tile (4 x 16B per thread)
  CFrag c;
  #pragma unroll
  for (int v = 0; v < 8; ++v) c.f[v] = 0.0f;

  auto issue_tile = [&](int j0, int buf) {
    // adj tile: 16 x 64 ints = 4 KB -> one b128 per thread
    async_copy_b128(lds_off(&adjt[buf][em][ej]),
                    adj + (size_t)(i0 + em) * N_NODES + j0 + ej);
    // Wh bf16 tile: 64 x 128 = 16 KB -> four b128 per thread
    #pragma unroll
    for (int q = 0; q < 4; ++q) {
      int id = q * 256 + threadIdx.x;       // chunk id 0..1023
      int r  = id >> 4;                     // tile row 0..63
      int cc = (id & 15) * 8;               // col (bf16 units)
      async_copy_b128(lds_off(&whbt[buf][r][cc]),
                      Whb + (size_t)(j0 + r) * F_OUT + cc);
    }
  };

  issue_tile(0, 0);                         // prime the pipeline (5 async loads)
  const int NT = N_NODES / JT;

  for (int t = 0; t < NT; ++t) {
    const int buf = t & 1;
    if (t + 1 < NT) {
      issue_tile((t + 1) * JT, buf ^ 1);    // overlap next tile with compute
      asm volatile("s_wait_asynccnt 5" ::: "memory");   // tile t complete
    } else {
      asm volatile("s_wait_asynccnt 0" ::: "memory");
    }
    __syncthreads();                        // tile t visible to all waves

    // ---- phase A: masked leaky-relu scores (registers) + partial row max ----
    int4   ad = *(const int4*)&adjt[buf][em][ej];
    float4 w2 = *(const float4*)(Wh2 + t * JT + ej);
    float e0 = wh1_m + w2.x, e1 = wh1_m + w2.y;
    float e2 = wh1_m + w2.z, e3 = wh1_m + w2.w;
    e0 = (e0 > 0.0f) ? e0 : ALPHA * e0;  e1 = (e1 > 0.0f) ? e1 : ALPHA * e1;
    e2 = (e2 > 0.0f) ? e2 : ALPHA * e2;  e3 = (e3 > 0.0f) ? e3 : ALPHA * e3;
    e0 = (ad.x > 0) ? e0 : MASK_VAL;     e1 = (ad.y > 0) ? e1 : MASK_VAL;
    e2 = (ad.z > 0) ? e2 : MASK_VAL;     e3 = (ad.w > 0) ? e3 : MASK_VAL;
    pmax[em][threadIdx.x & 15] = fmaxf(fmaxf(e0, e1), fmaxf(e2, e3));
    __syncthreads();

    // ---- phase B: running row max + rescale factor (wave 0, lanes 0-15) ----
    if (wv == 0 && lane < 16) {
      float mx = -3.0e38f;
      #pragma unroll
      for (int j = 0; j < 16; ++j) mx = fmaxf(mx, pmax[lane][j]);
      float mo = mrow[lane];
      float mn = fmaxf(mo, mx);
      mnew_s[lane]  = mn;
      scale_s[lane] = __expf(mo - mn);
      rsum_s[lane]  = 0.0f;
      mrow[lane]    = mn;
    }
    __syncthreads();

    // ---- phase C: p = exp(e - m) -> bf16 tile + row sums; rescale acc ----
    {
      float mn = mnew_s[em];
      float p0 = __expf(e0 - mn), p1 = __expf(e1 - mn);
      float p2 = __expf(e2 - mn), p3 = __expf(e3 - mn);
      *(unsigned int*)&ptile[em][ej]     =
          (unsigned int)f2bf(p0) | ((unsigned int)f2bf(p1) << 16);
      *(unsigned int*)&ptile[em][ej + 2] =
          (unsigned int)f2bf(p2) | ((unsigned int)f2bf(p3) << 16);
      atomicAdd(&rsum_s[em], (p0 + p1) + (p2 + p3));   // ds_add_f32
    }
    #pragma unroll
    for (int v = 0; v < 8; ++v) c.f[v] *= scale_s[v + 8 * hi];
    __syncthreads();

    // ---- phase D: update normalizer ----
    if (wv == 0 && lane < 16) lrow[lane] = lrow[lane] * scale_s[lane] + rsum_s[lane];

    // ---- phase E: acc += p(16x64) @ whbt(64x16 per wave), two K halves ----
    #pragma unroll
    for (int half = 0; half < 2; ++half) {
      AFrag a, b;
      #pragma unroll
      for (int v = 0; v < 8; ++v) {
        int ka = ((v < 4) ? (2 * v + 8 * hi) : (16 + 2 * (v - 4) + 8 * hi))
                 + 32 * half;
        a.u[v] = *(const unsigned int*)&ptile[mL][ka];     // bf16 pair
        int kb = ((v < 4) ? (2 * v + 8 * hi) : (16 + 2 * (v - 4) + 8 * hi)) + 32 * half;
        int col = wv * 16 + mL;
        unsigned int b0 = whbt[buf][kb][col];
        unsigned int b1 = whbt[buf][kb + 1][col];
        b.u[v] = b0 | (b1 << 16);
      }
      c.v = __builtin_amdgcn_wmma_f32_16x16x32_bf16(false, a.v, false, b.v,
                                                    (short)0, c.v, false, false);
    }
    __syncthreads();                        // protect buffers/ptile for reuse
  }

  // ---- epilogue: normalize + ELU ----
  #pragma unroll
  for (int v = 0; v < 8; ++v) {
    int row = v + 8 * hi;
    float val = c.f[v] / lrow[row];
    val = (val > 0.0f) ? val : (__expf(val) - 1.0f);
    out[(size_t)(i0 + row) * F_OUT + wv * 16 + mL] = val;
  }
}

// ---------------------------------------------------------------------------
extern "C" void kernel_launch(void* const* d_in, const int* in_sizes, int n_in,
                              void* d_out, int out_size, void* d_ws, size_t ws_size,
                              hipStream_t stream) {
  const float* h   = (const float*)d_in[0];
  const int*   adj = (const int*)  d_in[1];
  const float* w   = (const float*)d_in[2];
  const float* a   = (const float*)d_in[3];
  float* out = (float*)d_out;

  char* ws = (char*)d_ws;
  float* Wh            = (float*)ws;                                          // 4 MB
  unsigned short* Whb  = (unsigned short*)(ws + (size_t)N_NODES * F_OUT * 4); // 2 MB
  float* Wh1           = (float*)(ws + (size_t)N_NODES * F_OUT * 6);          // 32 KB
  float* Wh2           = Wh1 + N_NODES;                                       // 32 KB

  hipLaunchKernelGGL(wh_gemm_kernel, dim3(N_NODES / 16), dim3(256), 0, stream,
                     h, w, Wh, Whb);
  hipLaunchKernelGGL(attvec_kernel, dim3(N_NODES / 256), dim3(256), 0, stream,
                     Wh, a, Wh1, Wh2);
  hipLaunchKernelGGL(gat_fused_kernel, dim3(N_NODES / 16), dim3(256), 0, stream,
                     adj, Whb, Wh1, Wh2, out);
}
